// CLT_67542655697694
// MI455X (gfx1250) — matrix operand order repaired and checked by reading, not compiled, not run on hardware
//
#include <hip/hip_runtime.h>
#include <hip/hip_bf16.h>

// CLT forward for MI455X (gfx1250, wave32, WMMA).
// Roofline: ~283 GFLOP vs ~600 MB traffic (~470 FLOP/B) -> compute bound.
// Path: stage fp32 operands -> bf16 in LDS, v_wmma_f32_16x16x32_bf16 with
// fp32 accumulation; intermediate feat kept bf16 in workspace (64 MB).

typedef __bf16 bf16_t;
typedef __attribute__((ext_vector_type(16))) __bf16 v16bf;
typedef __attribute__((ext_vector_type(8)))  __bf16 v8bf;
typedef __attribute__((ext_vector_type(4)))  __bf16 v4bf;
typedef __attribute__((ext_vector_type(8)))  float  v8f;

#define BM  128     // output rows per block (B dim)
#define BN  128     // output cols per block
#define BK  64      // inner-dim slab per LDS stage (2 WMMA k-steps)
#define BKP 72      // padded LDS stride: 72*2B = 144B = 9*16B (alignment kept)
#define NT  256     // 8 waves of 32

__device__ __forceinline__ v8f wmma_bf16(v16bf a, v16bf b, v8f c) {
  // (neg_a, A, neg_b, B, c_mod, C, reuse_a, reuse_b)
  return __builtin_amdgcn_wmma_f32_16x16x32_bf16(false, a, false, b,
                                                 (short)0, c, false, false);
}

// A fragment (16x32 bf16, M rows x K): ISA layout — lane half g=lane/16 holds
// K chunks [g*8, g*8+8) and [16+g*8, 16+g*8+8) of row M = lane%16.
// LDS tile is row-major [row][k], stride BKP.
__device__ __forceinline__ v16bf frag_a(const bf16_t* lds, int m0, int ks, int lane) {
  const int r = lane & 15, g = lane >> 4;
  const bf16_t* base = lds + (size_t)(m0 + r) * BKP + ks + g * 8;
  v8bf lo = *(const v8bf*)(base);        // K = ks + g*8 + [0,8)
  v8bf hi = *(const v8bf*)(base + 16);   // K = ks + 16 + g*8 + [0,8)
  return __builtin_shufflevector(lo, hi, 0,1,2,3,4,5,6,7,8,9,10,11,12,13,14,15);
}

// B fragment (32x16 bf16, K x N cols): lane half g holds inner K = ks + g*8 +
// [0,8) and ks + 16 + g*8 + [0,8) of column N = lane%16.  LDS tile stored TRANSPOSED: [col][k], stride BKP.
__device__ __forceinline__ v16bf frag_b(const bf16_t* lds, int n0, int ks, int lane) {
  const int c = lane & 15, g = lane >> 4;
  const bf16_t* base = lds + (size_t)(n0 + c) * BKP + ks + g * 8;
  v8bf lo = *(const v8bf*)(base);
  v8bf hi = *(const v8bf*)(base + 16);
  return __builtin_shufflevector(lo, hi, 0,1,2,3,4,5,6,7,8,9,10,11,12,13,14,15);
}

// One BK slab of MMA: each wave computes a 32x64 region = 2x4 tiles.
__device__ __forceinline__ void mma_slab(const bf16_t* sA, const bf16_t* sB,
                                         int wm0, int wn0, int lane, v8f acc[2][4]) {
#pragma unroll
  for (int ks = 0; ks < BK; ks += 32) {
    v16bf a0 = frag_a(sA, wm0,      ks, lane);
    v16bf a1 = frag_a(sA, wm0 + 16, ks, lane);
#pragma unroll
    for (int jj = 0; jj < 4; ++jj) {
      v16bf b = frag_b(sB, wn0 + jj * 16, ks, lane);
      acc[0][jj] = wmma_bf16(a0, b, acc[0][jj]);
      acc[1][jj] = wmma_bf16(a1, b, acc[1][jj]);
    }
  }
}

// ---------------- Encoder: feat = jumprelu(x @ W_enc + b_enc, exp(log_th)) ---
// grid: (B/BM, K/BN, N); feat stored bf16 as [n][b][k] in workspace.
__global__ __launch_bounds__(NT) void clt_encoder(
    const float* __restrict__ x,       // [B][N][D]
    const float* __restrict__ W_enc,   // [N][D][K]
    const float* __restrict__ b_enc,   // [N][K]
    const float* __restrict__ log_th,  // [N][K]
    bf16_t* __restrict__ feat,         // [N][B][K]  (ws)
    int Bsz, int Nn, int Dd, int Kk)
{
  __shared__ bf16_t sA[BM * BKP];
  __shared__ bf16_t sB[BN * BKP];

  const int n    = blockIdx.z;
  const int bm0  = blockIdx.x * BM;
  const int bn0  = blockIdx.y * BN;
  const int tid  = threadIdx.x;
  const int lane = tid & 31;
  const int wave = tid >> 5;
  const int wm0  = (wave & 3) * 32;   // 4 waves along M
  const int wn0  = (wave >> 2) * 64;  // 2 waves along N

  v8f acc[2][4];
#pragma unroll
  for (int i = 0; i < 2; ++i)
#pragma unroll
    for (int j = 0; j < 4; ++j) acc[i][j] = (v8f)0.0f;

  const float* Wn = W_enc + (size_t)n * Dd * Kk;

  for (int k0 = 0; k0 < Dd; k0 += BK) {
    // Stage A: x[bm0+rr][n][k0..k0+BK) -> bf16, row-major, float4 reads.
    for (int idx = tid; idx < BM * (BK / 4); idx += NT) {
      const int rr = idx / (BK / 4);
      const int ch = (idx % (BK / 4)) * 4;
      const float4 v = *(const float4*)&x[((size_t)(bm0 + rr) * Nn + n) * Dd + k0 + ch];
      v4bf t = {(bf16_t)v.x, (bf16_t)v.y, (bf16_t)v.z, (bf16_t)v.w};
      *(v4bf*)&sA[(size_t)rr * BKP + ch] = t;
    }
    // Stage B transposed: W_enc[n][k0+kk][bn0+cc] -> sB[cc][kk], float4 reads.
    for (int idx = tid; idx < BK * (BN / 4); idx += NT) {
      const int kk = idx / (BN / 4);
      const int c4 = (idx % (BN / 4)) * 4;
      const float4 w = *(const float4*)&Wn[(size_t)(k0 + kk) * Kk + bn0 + c4];
      sB[(size_t)(c4 + 0) * BKP + kk] = (bf16_t)w.x;
      sB[(size_t)(c4 + 1) * BKP + kk] = (bf16_t)w.y;
      sB[(size_t)(c4 + 2) * BKP + kk] = (bf16_t)w.z;
      sB[(size_t)(c4 + 3) * BKP + kk] = (bf16_t)w.w;
    }
    if (k0 + BK < Dd)  // pull next weight slab toward L2
      __builtin_prefetch(&Wn[(size_t)(k0 + BK) * Kk + bn0 + (tid & 127)], 0, 1);
    __syncthreads();
    mma_slab(sA, sB, wm0, wn0, lane, acc);
    __syncthreads();
  }

  // Epilogue: +bias, jumprelu(hp, exp(log_th)), store bf16 feat.
  // C layout: VGPR v -> M = v + 8*(lane/16); N = lane%16.
  const int g = lane >> 4, cl = lane & 15;
#pragma unroll
  for (int jj = 0; jj < 4; ++jj) {
    const int kcol = bn0 + wn0 + jj * 16 + cl;
    const float bias = b_enc[(size_t)n * Kk + kcol];
    const float thr  = __expf(log_th[(size_t)n * Kk + kcol]);
#pragma unroll
    for (int ii = 0; ii < 2; ++ii) {
#pragma unroll
      for (int v = 0; v < 8; ++v) {
        const int row = bm0 + wm0 + ii * 16 + g * 8 + v;
        const float hp = acc[ii][jj][v] + bias;
        const float f  = (hp > thr) ? hp : 0.0f;
        feat[((size_t)n * Bsz + row) * Kk + kcol] = (bf16_t)f;
      }
    }
  }
}

// ---------------- Decoder: out[:,j,:] = sum_{l<=j} feat_l @ W_dec[p(l,j)] + biases
// grid: (B/BM, D/BN, N).  p(l,j) = l*N - l*(l-1)/2 + (j-l)  (triu row-major).
__global__ __launch_bounds__(NT) void clt_decoder(
    const bf16_t* __restrict__ feat,   // [N][B][K]
    const float* __restrict__ W_dec,   // [36][K][D]
    const float* __restrict__ b_dec,   // [36][D]
    float* __restrict__ out,           // [B][N][D]
    int Bsz, int Nn, int Dd, int Kk)
{
  __shared__ bf16_t sA[BM * BKP];
  __shared__ bf16_t sB[BN * BKP];

  const int jt   = blockIdx.z;
  const int bm0  = blockIdx.x * BM;
  const int bn0  = blockIdx.y * BN;
  const int tid  = threadIdx.x;
  const int lane = tid & 31;
  const int wave = tid >> 5;
  const int wm0  = (wave & 3) * 32;
  const int wn0  = (wave >> 2) * 64;

  v8f acc[2][4];
#pragma unroll
  for (int i = 0; i < 2; ++i)
#pragma unroll
    for (int j = 0; j < 4; ++j) acc[i][j] = (v8f)0.0f;

  for (int l = 0; l <= jt; ++l) {
    const int p = l * Nn - (l * (l - 1)) / 2 + (jt - l);
    const bf16_t* A  = feat  + (size_t)l * Bsz * Kk;
    const float*  Wp = W_dec + (size_t)p * Kk * Dd;

    for (int k0 = 0; k0 < Kk; k0 += BK) {
      // Stage A: already bf16 — straight 16B copies.
      for (int idx = tid; idx < BM * (BK / 8); idx += NT) {
        const int rr = idx / (BK / 8);
        const int ch = (idx % (BK / 8)) * 8;
        *(uint4*)&sA[(size_t)rr * BKP + ch] =
            *(const uint4*)&A[(size_t)(bm0 + rr) * Kk + k0 + ch];
      }
      // Stage B transposed: W_dec[p][k0+kk][bn0+cc] -> sB[cc][kk].
      for (int idx = tid; idx < BK * (BN / 4); idx += NT) {
        const int kk = idx / (BN / 4);
        const int c4 = (idx % (BN / 4)) * 4;
        const float4 w = *(const float4*)&Wp[(size_t)(k0 + kk) * Dd + bn0 + c4];
        sB[(size_t)(c4 + 0) * BKP + kk] = (bf16_t)w.x;
        sB[(size_t)(c4 + 1) * BKP + kk] = (bf16_t)w.y;
        sB[(size_t)(c4 + 2) * BKP + kk] = (bf16_t)w.z;
        sB[(size_t)(c4 + 3) * BKP + kk] = (bf16_t)w.w;
      }
      if (k0 + BK < Kk)
        __builtin_prefetch(&Wp[(size_t)(k0 + BK) * Dd + bn0 + (tid & 127)], 0, 1);
      __syncthreads();
      mma_slab(sA, sB, wm0, wn0, lane, acc);
      __syncthreads();
    }
  }

  // Epilogue: add sum of decoder biases over l<=jt, store fp32 out.
  const int g = lane >> 4, cl = lane & 15;
#pragma unroll
  for (int jj = 0; jj < 4; ++jj) {
    const int dcol = bn0 + wn0 + jj * 16 + cl;
    float bs = 0.0f;
    for (int l = 0; l <= jt; ++l) {
      const int p = l * Nn - (l * (l - 1)) / 2 + (jt - l);
      bs += b_dec[(size_t)p * Dd + dcol];
    }
#pragma unroll
    for (int ii = 0; ii < 2; ++ii) {
#pragma unroll
      for (int v = 0; v < 8; ++v) {
        const int row = bm0 + wm0 + ii * 16 + g * 8 + v;
        out[((size_t)row * Nn + jt) * Dd + dcol] = acc[ii][jj][v] + bs;
      }
    }
  }
}

extern "C" void kernel_launch(void* const* d_in, const int* in_sizes, int n_in,
                              void* d_out, int out_size, void* d_ws, size_t ws_size,
                              hipStream_t stream) {
  const float* x      = (const float*)d_in[0];
  const float* W_enc  = (const float*)d_in[1];
  const float* b_enc  = (const float*)d_in[2];
  const float* log_th = (const float*)d_in[3];
  const float* W_dec  = (const float*)d_in[4];
  const float* b_dec  = (const float*)d_in[5];

  // Derive shapes: s1=N*D*K, s2=N*K, s4=Ndec*K*D, s5=Ndec*D.
  const long s0 = in_sizes[0], s1 = in_sizes[1], s2 = in_sizes[2];
  const long s4 = in_sizes[4], s5 = in_sizes[5];
  const int D = (int)(s1 / s2);        // 768
  const int K = (int)(s4 / s5);        // 4096
  const int N = (int)(s2 / K);         // 8
  const int B = (int)(s0 / ((long)N * D)); // 1024

  bf16_t* feat = (bf16_t*)d_ws;        // [N][B][K] bf16 = 64 MB

  dim3 ge(B / BM, K / BN, N);
  clt_encoder<<<ge, NT, 0, stream>>>(x, W_enc, b_enc, log_th, feat, B, N, D, K);

  dim3 gd(B / BM, D / BN, N);
  clt_decoder<<<gd, NT, 0, stream>>>(feat, W_dec, b_dec, (float*)d_out, B, N, D, K);
}
